// DisAttLayer_11458972745981
// MI455X (gfx1250) — compile-verified, matrix operand order checked by reading, not run on hardware
//
#include <hip/hip_runtime.h>

// Problem constants (match reference)
#define B_  4
#define S_  256
#define H_  8
#define P_  32
#define BS_ 16
#define CS_ 16
#define L0_ 96
#define L1_ 32
#define L2_ 16

typedef __attribute__((ext_vector_type(16))) __bf16 v16bf;
typedef __attribute__((ext_vector_type(8)))  float  v8f;

union Frag { v16bf v; uint4 q[2]; };

__device__ __forceinline__ float4 ld4(const float* p) { return *(const float4*)p; }

__device__ __forceinline__ float relu(float v) { return fmaxf(v, 0.f); }  // single v_max_num_f32

__device__ __forceinline__ void pack8(v16bf& f, int base, float4 a, float4 b) {
    f[base + 0] = (__bf16)a.x; f[base + 1] = (__bf16)a.y;
    f[base + 2] = (__bf16)a.z; f[base + 3] = (__bf16)a.w;
    f[base + 4] = (__bf16)b.x; f[base + 5] = (__bf16)b.y;
    f[base + 6] = (__bf16)b.z; f[base + 7] = (__bf16)b.w;
}

__global__ __launch_bounds__(256) void disatt_wmma_kernel(
    const float* __restrict__ attn,
    const int*   __restrict__ b_seq,
    const int*   __restrict__ c_seq,
    const float* __restrict__ e_pos,   // [2S, H, P]
    const float* __restrict__ e_bi,    // [N_MB+1, H, BS]
    const float* __restrict__ e_bj,
    const float* __restrict__ e_ci,    // [N_C+2, H, CS]
    const float* __restrict__ e_cj,
    const float* __restrict__ w1,      // [L0, L1, H]
    const float* __restrict__ w2,      // [L1, L2, H]
    const float* __restrict__ w3,      // [L2, H]
    float* __restrict__ out)           // [B, H, S, S]
{
    // Weights stored TRANSPOSED so B-fragment reads are contiguous 16B runs.
    __shared__ __bf16 sW1t[L1_][L0_];   // [k][d]
    __shared__ __bf16 sW2t[L2_][L1_];   // [l][k]
    __shared__ float  sW3[L2_];
    __shared__ __bf16 sE1[8][16][L1_];  // per-wave relu(e1), row-major [row][k]
    __shared__ float  sT [8][16][L2_];  // per-wave relu(e2)*w3

    const int tid  = threadIdx.x;
    const int lane = tid & 31;
    const int wid  = tid >> 5;
    const int half = lane >> 4;         // 0: lanes 0-15, 1: lanes 16-31
    const int l16  = lane & 15;

    const int m = blockIdx.x % S_;
    const int h = (blockIdx.x / S_) % H_;
    const int b = blockIdx.x / (S_ * H_);

    // ---- Stage this head's weights, transposed, into LDS (bf16) ----
    for (int i = tid; i < L0_ * L1_; i += 256) {
        int d = i / L1_, k = i % L1_;
        sW1t[k][d] = (__bf16)w1[(d * L1_ + k) * H_ + h];
    }
    for (int i = tid; i < L1_ * L2_; i += 256) {
        int k = i / L2_, l = i % L2_;
        sW2t[l][k] = (__bf16)w2[(k * L2_ + l) * H_ + h];
    }
    if (tid < L2_) sW3[tid] = w3[tid * H_ + h];
    __syncthreads();

    // ---- B fragments: each half = one contiguous 16B LDS read (ds_load_b128) ----
    Frag B1[3][2];
    #pragma unroll
    for (int kb = 0; kb < 3; ++kb)
        #pragma unroll
        for (int ng = 0; ng < 2; ++ng) {
            B1[kb][ng].q[0] = *(const uint4*)&sW1t[ng * 16 + l16][kb * 32 + half * 8];
            B1[kb][ng].q[1] = *(const uint4*)&sW1t[ng * 16 + l16][kb * 32 + 16 + half * 8];
        }
    Frag B2;
    B2.q[0] = *(const uint4*)&sW2t[l16][half * 8];
    B2.q[1] = *(const uint4*)&sW2t[l16][16 + half * 8];

    // ---- Row-(m) features: iteration-invariant, hoisted (4 coalesced b128 loads) ----
    const int bi_row = b_seq[b * S_ + m];
    const int ci_row = c_seq[b * S_ + m];
    const float* bip = e_bi + ((size_t)bi_row * H_ + h) * BS_ + half * 8;
    const float* cip = e_ci + ((size_t)ci_row * H_ + h) * CS_ + half * 8;
    const float4 bi0 = ld4(bip), bi1 = ld4(bip + 4);
    const float4 ci0 = ld4(cip), ci1 = ld4(cip + 4);

    const v8f vzero = {0.f, 0.f, 0.f, 0.f, 0.f, 0.f, 0.f, 0.f};

    #pragma unroll
    for (int it = 0; it < 2; ++it) {
        const int n0 = (wid + it * 8) * 16;
        const int n  = n0 + l16;        // this lane's row of the 16xK A tile

        // ---- Build A fragments directly in registers from coalesced global loads.
        // Concat segments line up exactly with the 8-element K-runs of the A layout:
        //   kb=0: K 0..31  -> pos[m-n+S]         (runs at half*8 and 16+half*8)
        //   kb=1: K 32..63 -> bi[m] | bj[n]
        //   kb=2: K 64..95 -> ci[m] | cj[n]
        const float* pp = e_pos + ((size_t)(m - n + S_) * H_ + h) * P_ + half * 8;
        Frag A0, A1, A2f;
        pack8(A0.v, 0, ld4(pp),      ld4(pp + 4));
        pack8(A0.v, 8, ld4(pp + 16), ld4(pp + 20));

        const int bj_row = b_seq[b * S_ + n];
        const int cj_row = c_seq[b * S_ + n];
        const float* bjp = e_bj + ((size_t)bj_row * H_ + h) * BS_ + half * 8;
        const float* cjp = e_cj + ((size_t)cj_row * H_ + h) * CS_ + half * 8;
        pack8(A1.v,  0, bi0,      bi1);
        pack8(A1.v,  8, ld4(bjp), ld4(bjp + 4));
        pack8(A2f.v, 0, ci0,      ci1);
        pack8(A2f.v, 8, ld4(cjp), ld4(cjp + 4));

        // ---- GEMM1: [16x96] x [96x32] -> f32, 6 WMMAs ----
        v8f acc1_0, acc1_1;
        acc1_0 = __builtin_amdgcn_wmma_f32_16x16x32_bf16(
            false, A0.v, false, B1[0][0].v, (short)0, vzero, false, false);
        acc1_1 = __builtin_amdgcn_wmma_f32_16x16x32_bf16(
            false, A0.v, false, B1[0][1].v, (short)0, vzero, false, false);
        acc1_0 = __builtin_amdgcn_wmma_f32_16x16x32_bf16(
            false, A1.v, false, B1[1][0].v, (short)0, acc1_0, false, false);
        acc1_1 = __builtin_amdgcn_wmma_f32_16x16x32_bf16(
            false, A1.v, false, B1[1][1].v, (short)0, acc1_1, false, false);
        acc1_0 = __builtin_amdgcn_wmma_f32_16x16x32_bf16(
            false, A2f.v, false, B1[2][0].v, (short)0, acc1_0, false, false);
        acc1_1 = __builtin_amdgcn_wmma_f32_16x16x32_bf16(
            false, A2f.v, false, B1[2][1].v, (short)0, acc1_1, false, false);

        // ---- ReLU + relayout C->A through LDS (unavoidable lane transpose) ----
        #pragma unroll
        for (int i = 0; i < 8; ++i) {
            sE1[wid][i + half * 8][l16]      = (__bf16)relu(acc1_0[i]);
            sE1[wid][i + half * 8][16 + l16] = (__bf16)relu(acc1_1[i]);
        }
        asm volatile("s_wait_dscnt 0" ::: "memory");

        // ---- GEMM2: [16x32] x [32x16] -> f32, 1 WMMA ----
        Frag A2;
        A2.q[0] = *(const uint4*)&sE1[wid][l16][half * 8];
        A2.q[1] = *(const uint4*)&sE1[wid][l16][16 + half * 8];
        v8f acc2 = __builtin_amdgcn_wmma_f32_16x16x32_bf16(
            false, A2.v, false, B2.v, (short)0, vzero, false, false);

        // ---- relu(e2) * w3 -> LDS, per-row reduce (4x ds_load_b128), fused add-out ----
        #pragma unroll
        for (int i = 0; i < 8; ++i)
            sT[wid][i + half * 8][l16] = relu(acc2[i]) * sW3[l16];
        asm volatile("s_wait_dscnt 0" ::: "memory");

        if (lane < 16) {
            const float4* tr = (const float4*)&sT[wid][lane][0];
            float4 t0 = tr[0], t1 = tr[1], t2 = tr[2], t3 = tr[3];
            float da = ((t0.x + t0.y) + (t0.z + t0.w)) + ((t1.x + t1.y) + (t1.z + t1.w))
                     + ((t2.x + t2.y) + (t2.z + t2.w)) + ((t3.x + t3.y) + (t3.z + t3.w));
            size_t o = (((size_t)b * H_ + h) * S_ + m) * S_ + (size_t)(n0 + lane);
            out[o] = attn[o] + da;
        }
        // DS ops are in-order per wave: next iteration's sE1 stores cannot pass these reads.
    }
}

extern "C" void kernel_launch(void* const* d_in, const int* in_sizes, int n_in,
                              void* d_out, int out_size, void* d_ws, size_t ws_size,
                              hipStream_t stream) {
    (void)in_sizes; (void)n_in; (void)out_size; (void)d_ws; (void)ws_size;
    dim3 grid(B_ * H_ * S_);   // one (b,h,m) per block
    dim3 block(256);           // 8 waves; each wave: two 16-row n-tiles
    disatt_wmma_kernel<<<grid, block, 0, stream>>>(
        (const float*)d_in[0],
        (const int*)d_in[1],
        (const int*)d_in[2],
        (const float*)d_in[3],
        (const float*)d_in[4],
        (const float*)d_in[5],
        (const float*)d_in[6],
        (const float*)d_in[7],
        (const float*)d_in[8],
        (const float*)d_in[9],
        (const float*)d_in[10],
        (float*)d_out);
}
